// RelativeTransformerEncoderLayer_71167608095171
// MI455X (gfx1250) — matrix operand order repaired and checked by reading, not compiled, not run on hardware
//
#include <hip/hip_runtime.h>
#include <hip/hip_bf16.h>

// ---------------- problem constants ----------------
constexpr int T_   = 2048;
constexpr int B_   = 2;
constexpr int D_   = 1024;
constexpr int NH_  = 8;           // heads per direction
constexpr int DH_  = 64;          // head dim
constexpr int HO_  = NH_ * DH_;   // 512
constexpr int QKV_ = 3 * HO_;     // 1536
constexpr int DFF_ = 4096;
constexpr int M_   = T_ * B_;     // 4096 token rows

typedef __attribute__((ext_vector_type(16))) _Float16 v16h;
typedef __attribute__((ext_vector_type(8)))  _Float16 h8;
typedef __attribute__((ext_vector_type(8)))  float    v8f;
typedef __attribute__((ext_vector_type(4)))  unsigned int u32x4;
typedef __attribute__((ext_vector_type(4)))  int      i32x4;
typedef __attribute__((ext_vector_type(8)))  int      i32x8;

#if __has_builtin(__builtin_amdgcn_tensor_load_to_lds)
#define USE_TDM 1
#else
#define USE_TDM 0
#endif

static __device__ __forceinline__ v8f wmma_f16(const v16h& a, const v16h& b, const v8f& c) {
    return __builtin_amdgcn_wmma_f32_16x16x32_f16(false, a, false, b, (short)0, c, false, false);
}

// Issue one 2D TDM tile load: tileRows x tileCols f16 elements, row stride = rowLen elements,
// into LDS at byte offset lds_off. Descriptor packing per CDNA5 ISA 8.3/8.4 (D# groups).
static __device__ __forceinline__ void tdm_load_2d(unsigned int lds_off, const _Float16* gptr,
                                                   int rowLen, int tileRows, int tileCols) {
#if USE_TDM
    const unsigned long long ga = (unsigned long long)(__UINTPTR_TYPE__)gptr;
    u32x4 g0;
    g0[0] = 1u;                                                   // count=1, user descriptor
    g0[1] = lds_off;                                              // lds_addr (bytes)
    g0[2] = (unsigned int)ga;                                     // global_addr[31:0]
    g0[3] = (unsigned int)((ga >> 32) & 0x01FFFFFFu) | (2u << 30); // addr[56:32] | type=2
    i32x8 g1;
    g1[0] = 1 << 16;                                              // data_size = 2 bytes
    g1[1] = (rowLen & 0xFFFF) << 16;                              // tensor_dim0[15:0]
    g1[2] = ((rowLen >> 16) & 0xFFFF) | ((tileRows & 0xFFFF) << 16); // dim0 hi | tensor_dim1 lo
    g1[3] = ((tileRows >> 16) & 0xFFFF) | ((tileCols & 0xFFFF) << 16); // dim1 hi | tile_dim0
    g1[4] = tileRows & 0xFFFF;                                    // tile_dim1 (tile_dim2 = 0)
    g1[5] = rowLen;                                               // tensor_dim0_stride[31:0]
    g1[6] = 0;                                                    // stride hi | dim1 stride lo
    g1[7] = 0;
    const i32x4 gz = {0, 0, 0, 0};
#if defined(__clang_major__) && __clang_major__ >= 23
    const i32x8 gz8 = {0, 0, 0, 0, 0, 0, 0, 0};
    __builtin_amdgcn_tensor_load_to_lds(g0, g1, gz, gz, gz8, 0);
#else
    __builtin_amdgcn_tensor_load_to_lds(g0, g1, gz, gz, 0);
#endif
#else
    (void)lds_off; (void)gptr; (void)rowLen; (void)tileRows; (void)tileCols;
#endif
}

// ---------------- fp32 -> fp16 convert ----------------
__global__ void cvt_f32_f16_kernel(const float* __restrict__ src, _Float16* __restrict__ dst, int n) {
    int i = blockIdx.x * 256 + threadIdx.x;
    if (i < n) dst[i] = (_Float16)src[i];
}

// ---------------- LayerNorm (f32 in, f16 out, optional T-flip of output row) ----------------
__global__ __launch_bounds__(256) void layernorm_f16_kernel(
    const float* __restrict__ x, const float* __restrict__ g, const float* __restrict__ beta,
    _Float16* __restrict__ out, int flipOut)
{
    __shared__ float red0[8], red1[8];
    const int row = blockIdx.x;              // t*B + b
    const int t = row / B_, b = row % B_;
    const float* xr = x + (size_t)row * D_;
    float s = 0.f, s2 = 0.f;
    for (int i = threadIdx.x; i < D_; i += 256) { float v = xr[i]; s += v; s2 += v * v; }
    for (int off = 1; off < 32; off <<= 1) { s += __shfl_xor(s, off, 32); s2 += __shfl_xor(s2, off, 32); }
    const int w = threadIdx.x >> 5, lane = threadIdx.x & 31;
    if (lane == 0) { red0[w] = s; red1[w] = s2; }
    __syncthreads();
    if (threadIdx.x == 0) {
        float a = 0.f, c = 0.f;
        for (int i = 0; i < 8; ++i) { a += red0[i]; c += red1[i]; }
        red0[0] = a; red1[0] = c;
    }
    __syncthreads();
    const float mean = red0[0] * (1.0f / D_);
    const float var  = red1[0] * (1.0f / D_) - mean * mean;
    const float rstd = rsqrtf(var + 1e-5f);
    const int tOut = flipOut ? (T_ - 1 - t) : t;
    _Float16* orow = out + (size_t)(tOut * B_ + b) * D_;
    for (int i = threadIdx.x; i < D_; i += 256)
        orow[i] = (_Float16)((xr[i] - mean) * rstd * g[i] + beta[i]);
}

// ---------------- LDS-tiled WMMA GEMM fed by the Tensor Data Mover ----------------
// C[M,N] = A[M,K] * W[N,K]^T (+bias)(+resid)(relu).
// Block = 256 threads = 8 waves; block tile 64(M) x 128(N); BK = 64, double buffered in LDS.
// Wave w -> m-tile (w&3), n-strip (w>>2); each wave does 16x64 of C (8 WMMAs per k-step).
// grid = (M/64, N/128).
constexpr int BM = 64, BN = 128, BK = 64;
constexpr int BUFH = BM * BK + BN * BK;        // halves per double-buffer slot

template <bool OUTF16, bool RELU, bool BIAS, bool RESID>
__global__ void __launch_bounds__(256) gemm_wmma_kernel(
    const _Float16* __restrict__ A, const _Float16* __restrict__ W,
    const float* __restrict__ bias, const float* __restrict__ resid,
    void* __restrict__ Cout, int M, int N, int K)
{
    __shared__ _Float16 smem[2 * BUFH];        // 48 KB

    const int lane = threadIdx.x & 31;
    const int wv   = threadIdx.x >> 5;
    const int grp  = lane >> 4;
    const int nj   = lane & 15;
    const int mt   = wv & 3;                  // 0..3
    const int nstr = wv >> 2;                 // 0..1
    const int m0   = blockIdx.x * BM + mt * 16;
    const int n0   = blockIdx.y * BN + nstr * 64;
    const int nk   = K / BK;

    const _Float16* gA = A + (size_t)blockIdx.x * BM * K;
    const _Float16* gW = W + (size_t)blockIdx.y * BN * K;

    // buffer base pointers / LDS byte offsets computed arithmetically (no pointer arrays:
    // arrays of LDS-derived pointers become unsupported addrspacecast static initializers)
    auto sAbuf = [&](int buf) -> _Float16* { return smem + buf * BUFH; };
    auto sWbuf = [&](int buf) -> _Float16* { return smem + buf * BUFH + BM * BK; };
    auto offA  = [&](int buf) -> unsigned int { return (unsigned int)(buf * BUFH * 2); };
    auto offW  = [&](int buf) -> unsigned int { return (unsigned int)((buf * BUFH + BM * BK) * 2); };

    auto stage = [&](int buf, int k0) {
#if USE_TDM
        if (wv == 0) {
            tdm_load_2d(offA(buf), gA + k0, K, BM, BK);
            tdm_load_2d(offW(buf), gW + k0, K, BN, BK);
        }
#else
        {   // cooperative fallback: 256 threads, b128 global loads -> LDS
            const int r = threadIdx.x >> 2, c = (threadIdx.x & 3) * 16;
            *(v16h*)(sAbuf(buf) + r * BK + c) = *(const v16h*)(gA + (size_t)r * K + k0 + c);
#pragma unroll
            for (int rr = 0; rr < 2; ++rr) {
                const int wr = rr * 64 + r;
                *(v16h*)(sWbuf(buf) + wr * BK + c) = *(const v16h*)(gW + (size_t)wr * K + k0 + c);
            }
        }
#endif
    };

    v8f acc[4] = {v8f{}, v8f{}, v8f{}, v8f{}};
    stage(0, 0);
    for (int kt = 0; kt < nk; ++kt) {
        const int buf = kt & 1;
#if USE_TDM
        if (wv == 0) __builtin_amdgcn_s_wait_tensorcnt(0);   // tile for `buf` landed
#endif
        __syncthreads();                                     // release consumers / recycle other buf
        if (kt + 1 < nk) stage(buf ^ 1, (kt + 1) * BK);

        const _Float16* at = sAbuf(buf);
        const _Float16* wt = sWbuf(buf);
#pragma unroll
        for (int ks = 0; ks < 2; ++ks) {                     // two K=32 sub-steps
            v16h af;
            {
                const _Float16* ar = at + (mt * 16 + nj) * BK + ks * 32;
                const int kb = grp * 16;
                const h8 a0 = *(const h8*)(ar + kb);
                const h8 a1 = *(const h8*)(ar + kb + 8);
#pragma unroll
                for (int i = 0; i < 8; ++i) { af[i] = a0[i]; af[i + 8] = a1[i]; }
            }
#pragma unroll
            for (int sub = 0; sub < 4; ++sub) {
                const _Float16* wr = wt + (nstr * 64 + sub * 16 + nj) * BK + ks * 32 + grp * 16;
                const v16h bf = *(const v16h*)wr;
                acc[sub] = wmma_f16(af, bf, acc[sub]);
            }
        }
        __syncthreads();                                     // done reading `buf` before overwrite
    }

#pragma unroll
    for (int sub = 0; sub < 4; ++sub) {
        const int col = n0 + sub * 16 + nj;
        const float bv = BIAS ? bias[col] : 0.f;
#pragma unroll
        for (int r = 0; r < 8; ++r) {
            const int row = m0 + r + 8 * grp;
            float v = acc[sub][r] + bv;
            if (RESID) v += resid[(size_t)row * N + col];
            if (RELU)  v = fmaxf(v, 0.f);
            const size_t idx = (size_t)row * N + col;
            if (OUTF16) ((_Float16*)Cout)[idx] = (_Float16)v;
            else        ((float*)Cout)[idx]    = v;
        }
    }
    (void)M;
}

// ---------------- fused relative attention (Transformer-XL rel_shift, causal mask) ----------------
// One wave handles one 16-query tile for one (batch, head). 4 waves per block.
// qkv: [T*B, 1536] f16 (q|k|v). rk: [T, NH*DH] f16. out: [T*B, 2*HO] f16 cat buffer.
__global__ void __launch_bounds__(128) rel_attn_kernel(
    const _Float16* __restrict__ qkv, const _Float16* __restrict__ rk,
    const float* __restrict__ rwb, const float* __restrict__ rrb,
    _Float16* __restrict__ out, int colOff, int flipOut)
{
    __shared__ _Float16 pstage[4][16 * 36];     // wave-private P staging, padded stride
    const int lane = threadIdx.x & 31;
    const int w    = threadIdx.x >> 5;
    const int grp  = lane >> 4;
    const int nj   = lane & 15;
    const int i0   = (blockIdx.x * 4 + w) * 16;
    const int bn   = blockIdx.y;
    const int b    = bn / NH_;
    const int n    = bn % NH_;
    _Float16* lds  = pstage[w];

    // ---- Q fragments: (q + r_w_bias) for AC, (q + r_r_bias) for BD; K=64 -> two K=32 frags each
    v16h qa0, qa1, qr0, qr1;
    {
        const int t = i0 + nj;
        const _Float16* qrow = qkv + (size_t)(t * B_ + b) * QKV_ + n * DH_;
        const float* rwbn = rwb + n * DH_;
        const float* rrbn = rrb + n * DH_;
        const int kb = grp * 16;
#pragma unroll
        for (int i = 0; i < 8; ++i) {
            const int d0 = kb + i, d1 = kb + 8 + i;
            const float q0 = (float)qrow[d0], q1 = (float)qrow[d1];
            qa0[i] = (_Float16)(q0 + rwbn[d0]);  qa0[i + 8] = (_Float16)(q1 + rwbn[d1]);
            qr0[i] = (_Float16)(q0 + rrbn[d0]);  qr0[i + 8] = (_Float16)(q1 + rrbn[d1]);
            const int e0 = 32 + d0, e1 = 32 + d1;
            const float q2 = (float)qrow[e0], q3 = (float)qrow[e1];
            qa1[i] = (_Float16)(q2 + rwbn[e0]);  qa1[i + 8] = (_Float16)(q3 + rwbn[e1]);
            qr1[i] = (_Float16)(q2 + rrbn[e0]);  qr1[i + 8] = (_Float16)(q3 + rrbn[e1]);
        }
    }

    v8f pv[4] = {v8f{}, v8f{}, v8f{}, v8f{}};
    float mrow[8], lrow[8];
#pragma unroll
    for (int r = 0; r < 8; ++r) { mrow[r] = -1e30f; lrow[r] = 0.f; }

    const int jEnd = i0 + 15;                   // causal: keys j <= i
    for (int j0 = 0; j0 <= jEnd; j0 += 32) {
        v8f s[2];
#pragma unroll
        for (int half = 0; half < 2; ++half) {
            const int js = j0 + half * 16;
            // AC = (q+rwb) . K^T  : 2 WMMAs over K=64
            v8f ac = v8f{};
            {
                int krow = js + nj; if (krow > T_ - 1) krow = T_ - 1;
                const _Float16* kr = qkv + (size_t)(krow * B_ + b) * QKV_ + HO_ + n * DH_ + grp * 16;
                const v16h kb0 = *(const v16h*)kr;
                const v16h kb1 = *(const v16h*)(kr + 32);
                ac = wmma_f16(qa0, kb0, ac);
                ac = wmma_f16(qa1, kb1, ac);
            }
            // BD raw band: out[i,j] = raw[i, T-1 + j - i]; need 31 r-values -> two 16x16 tiles
            const int rb = T_ - 1 + js - i0 - 15;
            v8f bd0 = v8f{}, bd1 = v8f{};
            {
                int r0 = rb + nj;       r0 = r0 < 0 ? 0 : (r0 > T_ - 1 ? T_ - 1 : r0);
                int r1 = rb + 16 + nj;  r1 = r1 < 0 ? 0 : (r1 > T_ - 1 ? T_ - 1 : r1);
                const _Float16* p0 = rk + (size_t)r0 * HO_ + n * DH_ + grp * 16;
                const _Float16* p1 = rk + (size_t)r1 * HO_ + n * DH_ + grp * 16;
                const v16h rb00 = *(const v16h*)p0;
                const v16h rb01 = *(const v16h*)(p0 + 32);
                const v16h rb10 = *(const v16h*)p1;
                const v16h rb11 = *(const v16h*)(p1 + 32);
                bd0 = wmma_f16(qr0, rb00, bd0);  bd0 = wmma_f16(qr1, rb01, bd0);
                bd1 = wmma_f16(qr0, rb10, bd1);  bd1 = wmma_f16(qr1, rb11, bd1);
            }
            // shifted gather + combine + mask
            v8f sc;
#pragma unroll
            for (int r = 0; r < 8; ++r) {
                const int mi = r + 8 * grp;
                const int c  = 15 + nj - mi;                 // 0..30
                const int sl = grp * 16 + (c & 15);
                const float g0 = __shfl(bd0[r], sl, 32);
                const float g1 = __shfl(bd1[r], sl, 32);
                const float bd = (c < 16) ? g0 : g1;
                float v = (ac[r] + bd) * 0.125f;             // 1/sqrt(64)
                if (js + nj > i0 + mi) v = -1e30f;           // causal mask (True = masked)
                sc[r] = v;
            }
            s[half] = sc;
        }

        // ---- online softmax over the 32 new keys ----
#pragma unroll
        for (int r = 0; r < 8; ++r) {
            float v = fmaxf(s[0][r], s[1][r]);
#pragma unroll
            for (int off = 1; off < 16; off <<= 1) v = fmaxf(v, __shfl_xor(v, off, 32));
            const float mnew  = fmaxf(mrow[r], v);
            const float scale = __expf(mrow[r] - mnew);
            const float p0 = __expf(s[0][r] - mnew);
            const float p1 = __expf(s[1][r] - mnew);
            float ls = p0 + p1;
#pragma unroll
            for (int off = 1; off < 16; off <<= 1) ls += __shfl_xor(ls, off, 32);
            lrow[r] = lrow[r] * scale + ls;
            mrow[r] = mnew;
#pragma unroll
            for (int sub = 0; sub < 4; ++sub) pv[sub][r] *= scale;
            const int mi = r + 8 * grp;
            lds[mi * 36 + nj]      = (_Float16)p0;
            lds[mi * 36 + 16 + nj] = (_Float16)p1;
        }
        // wave-private LDS staging: DS ops are in-order per wave; wait them out explicitly
        asm volatile("s_wait_dscnt 0x0" ::: "memory");

        // ---- P (16x32) x V (32x64) : 4 WMMAs ----
        v16h pa;
        {
            const int kb = grp * 16;
#pragma unroll
            for (int i = 0; i < 8; ++i) {
                pa[i]     = lds[nj * 36 + kb + i];
                pa[i + 8] = lds[nj * 36 + kb + 8 + i];
            }
        }
#pragma unroll
        for (int sub = 0; sub < 4; ++sub) {
            v16h vb;
#pragma unroll
            for (int e = 0; e < 16; ++e) {
                const int kk = grp * 16 + e;
                int krow = j0 + kk; if (krow > T_ - 1) krow = T_ - 1;
                vb[e] = qkv[(size_t)(krow * B_ + b) * QKV_ + 2 * HO_ + n * DH_ + sub * 16 + nj];
            }
            pv[sub] = wmma_f16(pa, vb, pv[sub]);
        }
    }

    // ---- finalize + store into concat buffer (bwd direction writes flipped rows, col offset) ----
#pragma unroll
    for (int sub = 0; sub < 4; ++sub) {
#pragma unroll
        for (int r = 0; r < 8; ++r) {
            const int mi = r + 8 * grp;
            const int t  = i0 + mi;
            const int tOut = flipOut ? (T_ - 1 - t) : t;
            const float val = pv[sub][r] / lrow[r];
            out[(size_t)(tOut * B_ + b) * (2 * HO_) + colOff + n * DH_ + sub * 16 + nj] = (_Float16)val;
        }
    }
}

// ---------------- host-side orchestration ----------------
extern "C" void kernel_launch(void* const* d_in, const int* in_sizes, int n_in,
                              void* d_out, int out_size, void* d_ws, size_t ws_size,
                              hipStream_t stream) {
    (void)in_sizes; (void)n_in; (void)out_size; (void)ws_size;
    const float* x      = (const float*)d_in[0];
    const float* pos    = (const float*)d_in[1];
    // d_in[2], d_in[3]: masks (both causal; hardcoded in the attention kernel)
    const float* ln1_g  = (const float*)d_in[4];
    const float* ln1_b  = (const float*)d_in[5];
    const float* ln2_g  = (const float*)d_in[6];
    const float* ln2_b  = (const float*)d_in[7];
    const float* ln3_g  = (const float*)d_in[8];
    const float* ln3_b  = (const float*)d_in[9];
    const float* qkv_f  = (const float*)d_in[10];
    const float* rnet_f = (const float*)d_in[11];
    const float* rwb_f  = (const float*)d_in[12];
    const float* rrb_f  = (const float*)d_in[13];
    const float* qkv_b  = (const float*)d_in[14];
    const float* rnet_b = (const float*)d_in[15];
    const float* rwb_b  = (const float*)d_in[16];
    const float* rrb_b  = (const float*)d_in[17];
    const float* wo     = (const float*)d_in[18];
    const float* bo     = (const float*)d_in[19];
    const float* w1     = (const float*)d_in[20];
    const float* b1     = (const float*)d_in[21];
    const float* w2     = (const float*)d_in[22];
    const float* b2     = (const float*)d_in[23];
    float* outp = (float*)d_out;

    // ---- workspace carve-up (256B aligned) ----
    char* ws = (char*)d_ws;
    size_t off = 0;
    auto a16 = [&](size_t n) { _Float16* p = (_Float16*)(ws + off); off += ((n * 2 + 255) / 256) * 256; return p; };
    auto a32 = [&](size_t n) { float*    p = (float*)(ws + off);    off += ((n * 4 + 255) / 256) * 256; return p; };

    _Float16* wqf  = a16((size_t)QKV_ * D_);
    _Float16* wqb  = a16((size_t)QKV_ * D_);
    _Float16* wrf  = a16((size_t)HO_ * D_);
    _Float16* wrb  = a16((size_t)HO_ * D_);
    _Float16* wwo  = a16((size_t)D_ * D_);
    _Float16* ww1  = a16((size_t)DFF_ * D_);
    _Float16* ww2  = a16((size_t)D_ * DFF_);
    _Float16* posh = a16((size_t)T_ * D_);
    _Float16* ln1o = a16((size_t)M_ * D_);
    _Float16* ln2o = a16((size_t)M_ * D_);
    _Float16* qkvF = a16((size_t)M_ * QKV_);
    _Float16* qkvB = a16((size_t)M_ * QKV_);
    _Float16* rkF  = a16((size_t)T_ * HO_);
    _Float16* rkB  = a16((size_t)T_ * HO_);
    _Float16* cat  = a16((size_t)M_ * D_);
    float*    hbuf = a32((size_t)M_ * D_);
    _Float16* ln3o = a16((size_t)M_ * D_);
    _Float16* y1   = a16((size_t)M_ * DFF_);

    auto cvt = [&](const float* s, _Float16* d, int n) {
        cvt_f32_f16_kernel<<<(n + 255) / 256, 256, 0, stream>>>(s, d, n);
    };
    // 1) weight + pos conversion
    cvt(qkv_f,  wqf,  QKV_ * D_);
    cvt(qkv_b,  wqb,  QKV_ * D_);
    cvt(rnet_f, wrf,  HO_ * D_);
    cvt(rnet_b, wrb,  HO_ * D_);
    cvt(wo,     wwo,  D_ * D_);
    cvt(w1,     ww1,  DFF_ * D_);
    cvt(w2,     ww2,  D_ * DFF_);
    cvt(pos,    posh, T_ * D_);

    // 2) layernorms (LN2 writes flipped rows == LN of flipped input)
    layernorm_f16_kernel<<<M_, 256, 0, stream>>>(x, ln1_g, ln1_b, ln1o, 0);
    layernorm_f16_kernel<<<M_, 256, 0, stream>>>(x, ln2_g, ln2_b, ln2o, 1);

    // 3) QKV projections (M=4096, N=1536, K=1024)
    gemm_wmma_kernel<true, false, false, false><<<dim3(M_ / BM, QKV_ / BN), 256, 0, stream>>>(
        ln1o, wqf, nullptr, nullptr, qkvF, M_, QKV_, D_);
    gemm_wmma_kernel<true, false, false, false><<<dim3(M_ / BM, QKV_ / BN), 256, 0, stream>>>(
        ln2o, wqb, nullptr, nullptr, qkvB, M_, QKV_, D_);

    // 4) relative-position key projections (M=2048, N=512, K=1024)
    gemm_wmma_kernel<true, false, false, false><<<dim3(T_ / BM, HO_ / BN), 256, 0, stream>>>(
        posh, wrf, nullptr, nullptr, rkF, T_, HO_, D_);
    gemm_wmma_kernel<true, false, false, false><<<dim3(T_ / BM, HO_ / BN), 256, 0, stream>>>(
        posh, wrb, nullptr, nullptr, rkB, T_, HO_, D_);

    // 5) fused relative attention; bwd writes flipped rows into cols [512,1024)
    rel_attn_kernel<<<dim3(T_ / 64, B_ * NH_), 128, 0, stream>>>(qkvF, rkF, rwb_f, rrb_f, cat, 0,   0);
    rel_attn_kernel<<<dim3(T_ / 64, B_ * NH_), 128, 0, stream>>>(qkvB, rkB, rwb_b, rrb_b, cat, HO_, 1);

    // 6) output projection + bias + residual(x) -> h (f32)
    gemm_wmma_kernel<false, false, true, true><<<dim3(M_ / BM, D_ / BN), 256, 0, stream>>>(
        cat, wwo, bo, x, hbuf, M_, D_, D_);

    // 7) FFN: LN3 -> w1 (+b1, relu) -> w2 (+b2, +h) -> d_out
    layernorm_f16_kernel<<<M_, 256, 0, stream>>>(hbuf, ln3_g, ln3_b, ln3o, 0);
    gemm_wmma_kernel<true, true, true, false><<<dim3(M_ / BM, DFF_ / BN), 256, 0, stream>>>(
        ln3o, ww1, b1, nullptr, y1, M_, DFF_, D_);
    gemm_wmma_kernel<false, false, true, true><<<dim3(M_ / BM, D_ / BN), 256, 0, stream>>>(
        y1, ww2, b2, hbuf, outp, M_, D_, DFF_);
}
